// EmaAttention_26731876450900
// MI455X (gfx1250) — compile-verified, matrix operand order checked by reading, not run on hardware
//
#include <hip/hip_runtime.h>
#include <hip/hip_bf16.h>

// ---------------------------------------------------------------------------
// EMA-MEGA attention for MI455X (gfx1250, wave32, WMMA bf16 + TDM).
// Pipeline: EMA scan (VALU) -> QKV GEMM (WMMA, TDM-staged A) -> fused
// attention with post-softmax Toeplitz bias (WMMA, TDM-staged K) -> out-proj.
// ---------------------------------------------------------------------------

#define BB 32
#define NN 577          // N = NP + 1
#define LL 576          // NP
#define DD 512          // DIM
#define HH 8            // HEADS
#define DHD 64          // DH
#define NDIM 16
#define MROWS (BB * NN)         // 18464  (divisible by 16)
#define NQKV (3 * DD)           // 1536
#define JT_TOT 37               // ceil(577/16)
#define JPAIRS 19               // ceil(38/2), tile 37 fully masked
#define VT_STRIDE 640           // padded j length for V^T rows (mult of 32)

typedef unsigned short u16;
typedef float v8f __attribute__((ext_vector_type(8)));
typedef __bf16 bf16x16 __attribute__((ext_vector_type(16)));
typedef unsigned int u32x4 __attribute__((ext_vector_type(4)));
typedef int i32x4 __attribute__((ext_vector_type(4)));
typedef int i32x8 __attribute__((ext_vector_type(8)));

union Frag {                    // one 16x32 bf16 A/B WMMA operand per lane
    bf16x16 v;
    uint4 u[2];
    u16 s[16];
};

__device__ __forceinline__ u16 f2bf(float f) {
    unsigned u = __float_as_uint(f);
    unsigned r = u + 0x7FFFu + ((u >> 16) & 1u);   // round-to-nearest-even
    return (u16)(r >> 16);
}

// ---------------------------------------------------------------------------
// Tensor Data Mover: 2-D tile of 2-byte elements, global -> LDS.
// D# bit packing per CDNA5 ISA ch.8 (group0: count/lds_addr/global_addr/type,
// group1: data_size/tensor dims/tile dims/stride). Rows/cols beyond
// tensor_dim* are hardware zero-filled (used as the M-guard / j-padding).
// ---------------------------------------------------------------------------
__device__ __forceinline__ void tdm_load_2d_b16(void* lds_dst, const void* gsrc,
                                                unsigned tensor_d0, unsigned tensor_d1,
                                                unsigned tile_d0, unsigned tile_d1,
                                                unsigned long long stride_elems) {
    unsigned long long ga = (unsigned long long)gsrc;
    u32x4 g0;
    g0[0] = 1u;                                            // count=1 (valid, user mode)
    g0[1] = (unsigned)(unsigned long long)lds_dst;         // LDS byte address (addr[31:0])
    g0[2] = (unsigned)(ga & 0xFFFFFFFFull);                // global_addr[31:0]
    g0[3] = (unsigned)((ga >> 32) & 0x01FFFFFFull) | (2u << 30);  // ga[56:32] | type=2

    i32x8 g1;
    g1[0] = 0x00010000;                                    // wg_mask=0, data_size=2B
    g1[1] = (int)(tensor_d0 << 16);                        // tensor_dim0[15:0]
    g1[2] = (int)((tensor_d0 >> 16) | (tensor_d1 << 16));  // td0[31:16] | td1[15:0]
    g1[3] = (int)((tensor_d1 >> 16) | (tile_d0 << 16));    // td1[31:16] | tile_dim0
    g1[4] = (int)(tile_d1 & 0xFFFFu);                      // tile_dim1, tile_dim2=0
    g1[5] = (int)(stride_elems & 0xFFFFFFFFull);           // tensor_dim0_stride[31:0]
    g1[6] = (int)((stride_elems >> 32) & 0xFFFFull);       // stride[47:32], dim1_stride=0
    g1[7] = 0;

    i32x4 z4 = {};
#if __clang_major__ >= 23
    i32x8 z8 = {};
    __builtin_amdgcn_tensor_load_to_lds(g0, g1, z4, z4, z8, 0);
#else
    __builtin_amdgcn_tensor_load_to_lds(g0, g1, z4, z4, 0);
#endif
}

// ---------------------------------------------------------------------------
// Weight prep: W_all = [W_qk | W_v] -> bf16 (512 x 1536), W_out -> bf16.
// ---------------------------------------------------------------------------
__global__ void prep_wall_kernel(const float* __restrict__ Wqk,
                                 const float* __restrict__ Wv,
                                 u16* __restrict__ wall) {
    int idx = blockIdx.x * blockDim.x + threadIdx.x;
    if (idx >= DD * NQKV) return;
    int k = idx / NQKV, c = idx % NQKV;
    float v = (c < 1024) ? Wqk[k * 1024 + c] : Wv[k * DD + (c - 1024)];
    wall[idx] = f2bf(v);
}

__global__ void prep_wout_kernel(const float* __restrict__ Wout,
                                 u16* __restrict__ wout) {
    int idx = blockIdx.x * blockDim.x + threadIdx.x;
    if (idx >= DD * DD) return;
    wout[idx] = f2bf(Wout[idx]);
}

// cls token row: xm[b*N + 0] = bf16(x[b, 0, :])
__global__ void prep_cls_kernel(const float* __restrict__ x,
                                u16* __restrict__ xm) {
    int idx = blockIdx.x * blockDim.x + threadIdx.x;
    if (idx >= BB * DD) return;
    int b = idx >> 9, d = idx & 511;
    size_t off = (size_t)b * NN * DD + d;
    xm[off] = f2bf(x[off]);
}

// ---------------------------------------------------------------------------
// EMA forward scan: 16-state recurrence per (b, d), writes y_fwd to scratch.
// ---------------------------------------------------------------------------
__global__ void __launch_bounds__(256)
ema_fwd_kernel(const float* __restrict__ x,
               const float* __restrict__ edelta, const float* __restrict__ ealpha,
               const float* __restrict__ ebeta,  const float* __restrict__ egamma,
               float* __restrict__ yfwd) {
    int idx = blockIdx.x * blockDim.x + threadIdx.x;
    if (idx >= BB * DD) return;
    int b = idx >> 9, d = idx & 511;

    float q[NDIM], w[NDIM], s[NDIM];
#pragma unroll
    for (int n = 0; n < NDIM; ++n) {
        int pi = d * NDIM + n;
        float p  = 1.0f / (1.0f + __expf(-edelta[pi]));
        float sa = 1.0f / (1.0f + __expf(-ealpha[pi]));
        q[n] = 1.0f - p * sa;
        w[n] = p * ebeta[pi] * egamma[pi] * 0.25f;   // scale = sqrt(1/16)
        s[n] = 0.0f;
    }
    for (int t = 0; t < LL; ++t) {
        float xv = x[((size_t)b * NN + 1 + t) * DD + d];
        float y = 0.0f;
#pragma unroll
        for (int n = 0; n < NDIM; ++n) { s[n] = fmaf(q[n], s[n], xv); y = fmaf(w[n], s[n], y); }
        yfwd[((size_t)b * LL + t) * DD + d] = y;
    }
}

// EMA backward scan + silu finalize -> xm bf16 rows b*N + 1 + t.
__global__ void __launch_bounds__(256)
ema_bwd_kernel(const float* __restrict__ x,
               const float* __restrict__ edelta, const float* __restrict__ ealpha,
               const float* __restrict__ ebeta,  const float* __restrict__ egamma,
               const float* __restrict__ eomega,
               const float* __restrict__ yfwd,
               u16* __restrict__ xm) {
    int idx = blockIdx.x * blockDim.x + threadIdx.x;
    if (idx >= BB * DD) return;
    int b = idx >> 9, d = idx & 511;

    float q[NDIM], w[NDIM], s[NDIM];
#pragma unroll
    for (int n = 0; n < NDIM; ++n) {
        int pi = (DD + d) * NDIM + n;                // backward params at d+512
        float p  = 1.0f / (1.0f + __expf(-edelta[pi]));
        float sa = 1.0f / (1.0f + __expf(-ealpha[pi]));
        q[n] = 1.0f - p * sa;
        w[n] = p * ebeta[pi] * egamma[pi] * 0.25f;
        s[n] = 0.0f;
    }
    float om = eomega[d];
    for (int tt = 0; tt < LL; ++tt) {
        int t = LL - 1 - tt;
        float xv = x[((size_t)b * NN + 1 + t) * DD + d];
        float y = 0.0f;
#pragma unroll
        for (int n = 0; n < NDIM; ++n) { s[n] = fmaf(q[n], s[n], xv); y = fmaf(w[n], s[n], y); }
        float tot = yfwd[((size_t)b * LL + t) * DD + d] + y + xv * om;
        float mv = tot / (1.0f + __expf(-tot));      // silu
        xm[((size_t)b * NN + 1 + t) * DD + d] = f2bf(mv);
    }
}

// ---------------------------------------------------------------------------
// Generic 128x128 WMMA bf16 GEMM, K stepped by 32. 8 waves in 2(M)x4(N) grid,
// each wave owns 4x2 = 8 accumulator tiles (64x32).
// A tile is staged via TDM (one 16x32 slice per wave, M-guard via OOB
// zero-fill); B tile is staged transposed by VALU (TDM cannot transpose).
// MODE 0: bf16 epilogue, columns < qcols scaled by 0.125 (q pre-scale).
// MODE 1: f32 epilogue + bias (final projection into d_out).
// ---------------------------------------------------------------------------
template <int MODE>
__global__ void __launch_bounds__(256)
gemm_wmma_kernel(const u16* __restrict__ A,   // M x 512, bf16 row-major
                 const u16* __restrict__ Bg,  // 512 x Nd, bf16 row-major
                 u16* __restrict__ Cbf,
                 float* __restrict__ Cf, const float* __restrict__ bias,
                 int M, int Nd, int qcols) {
    __shared__ u16 As[128 * 32];     // (m, k)
    __shared__ u16 Bs[128 * 32];     // (n, k) : staged transposed

    const int tid = threadIdx.x;
    const int lane = tid & 31, wave = tid >> 5;
    const int lr = lane & 15, half = (lane >> 4) & 1;
    const int wm = wave >> 2, wn = wave & 3;
    const int bm0 = blockIdx.x * 128, bn0 = blockIdx.y * 128;

    const int arow0 = bm0 + wave * 16;               // this wave's A slice
    const int arem  = M - arow0;
    const unsigned atd1 = (arem > 0) ? (unsigned)arem : 0u;   // rows valid; rest zero-fill

    v8f acc[4][2];
#pragma unroll
    for (int i = 0; i < 4; ++i)
#pragma unroll
        for (int j = 0; j < 2; ++j) acc[i][j] = (v8f){};

    for (int kk = 0; kk < 16; ++kk) {
        const int k0 = kk * 32;
        __syncthreads();
        // --- stage A slice (16 rows x 32 cols) via Tensor Data Mover ---
        tdm_load_2d_b16(&As[wave * 16 * 32], &A[(size_t)arow0 * 512 + k0],
                        /*tensor_d0=*/32u, /*tensor_d1=*/atd1,
                        /*tile_d0=*/32u, /*tile_d1=*/16u, /*stride=*/512ull);
        // --- stage B tile transposed -> (n, k), overlaps with TDM ---
        for (int c = tid; c < 512; c += 256) {
            int kr = c >> 4, part = c & 15;
            int gk = k0 + kr, gn = bn0 + part * 8;
            uint4 v = *(const uint4*)&Bg[(size_t)gk * Nd + gn];
            if (kk < 15) __builtin_prefetch(&Bg[(size_t)(gk + 32) * Nd + gn], 0, 0);
            const u16* pv = (const u16*)&v;
#pragma unroll
            for (int e = 0; e < 8; ++e) Bs[(part * 8 + e) * 32 + kr] = pv[e];
        }
        __builtin_amdgcn_s_wait_tensorcnt(0);
        __syncthreads();

        Frag af[4], bf[2];
#pragma unroll
        for (int mt = 0; mt < 4; ++mt) {
            const u16* p = &As[(wm * 64 + mt * 16 + lr) * 32 + half * 8];
            af[mt].u[0] = *(const uint4*)p;
            af[mt].u[1] = *(const uint4*)(p + 16);
        }
#pragma unroll
        for (int nt = 0; nt < 2; ++nt) {
            const u16* p = &Bs[(wn * 32 + nt * 16 + lr) * 32 + half * 8];
            bf[nt].u[0] = *(const uint4*)p;
            bf[nt].u[1] = *(const uint4*)(p + 16);
        }
#pragma unroll
        for (int mt = 0; mt < 4; ++mt)
#pragma unroll
            for (int nt = 0; nt < 2; ++nt)
                acc[mt][nt] = __builtin_amdgcn_wmma_f32_16x16x32_bf16(
                    false, af[mt].v, false, bf[nt].v, (short)0, acc[mt][nt], false, false);
    }

    // --- epilogue ---
#pragma unroll
    for (int mt = 0; mt < 4; ++mt)
#pragma unroll
        for (int nt = 0; nt < 2; ++nt) {
            int n = bn0 + wn * 32 + nt * 16 + lr;
            int mb = bm0 + wm * 64 + mt * 16 + half * 8;
#pragma unroll
            for (int r = 0; r < 8; ++r) {
                int m = mb + r;
                if (m < M) {
                    float v = acc[mt][nt][r];
                    if (MODE == 0) {
                        if (n < qcols) v *= 0.125f;      // DH^-0.5 folded into q
                        Cbf[(size_t)m * Nd + n] = f2bf(v);
                    } else {
                        Cf[(size_t)m * Nd + n] = v + bias[n];
                    }
                }
            }
        }
}

// ---------------------------------------------------------------------------
// Fused attention per (b, h): S^T = K.Q^T (so lanes index i, VGPRs index j),
// two-pass softmax, post-softmax Toeplitz bias folded into a second WMMA
// accumulator, O^T = V^T . P^T straight out of the C-layout (no shuffles).
// K tile is staged by TDM (rows 577..591 hardware zero-filled).
// ---------------------------------------------------------------------------
__device__ __forceinline__ v8f st_tile(const u16* __restrict__ kl,
                                       const Frag* __restrict__ qf,
                                       int jt, int lr, int half) {
    v8f st = (v8f){};
#pragma unroll
    for (int ks = 0; ks < 2; ++ks) {
        Frag kf;
        const u16* p = &kl[(jt * 16 + lr) * DHD + ks * 32 + half * 8];
        kf.u[0] = *(const uint4*)p;
        kf.u[1] = *(const uint4*)(p + 16);
        st = __builtin_amdgcn_wmma_f32_16x16x32_bf16(
            false, kf.v, false, qf[ks].v, (short)0, st, false, false);
    }
    return st;
}

__global__ void __launch_bounds__(256)
attention_kernel(const u16* __restrict__ qkv,       // (B*N, 1536) bf16
                 const float* __restrict__ relb,    // (1153,)
                 u16* __restrict__ aout) {          // (B*N, 512) bf16
    extern __shared__ char smem[];
    u16*   kl  = (u16*)smem;                               // 592 x 64
    u16*   vT  = (u16*)(smem + 592 * DHD * 2);             // 64 x 640 (j-padded)
    float* rbs = (float*)(smem + 592 * DHD * 2 + DHD * VT_STRIDE * 2);

    const int b = blockIdx.x >> 3, h = blockIdx.x & 7;
    const int tid = threadIdx.x;
    const int lane = tid & 31, wave = tid >> 5;
    const int lr = lane & 15, half = (lane >> 4) & 1;
    const size_t rowbase = (size_t)b * NN;

    // ---- stage K via TDM: 592x64 tile of the (577 x 1536) qkv tensor.
    // tensor_dim1=577 with tile_dim1=592 -> rows 577..591 zero-filled by HW.
    if (wave == 0) {
        tdm_load_2d_b16(kl, &qkv[rowbase * NQKV + DD + h * DHD],
                        /*tensor_d0=*/64u, /*tensor_d1=*/(unsigned)NN,
                        /*tile_d0=*/64u, /*tile_d1=*/592u, /*stride=*/(unsigned long long)NQKV);
        __builtin_amdgcn_s_wait_tensorcnt(0);
    }
    // ---- stage rel bias + zero V^T pad while TDM runs ----
    for (int i = tid; i < 2 * NN - 1; i += 256) rbs[i] = relb[i];
    for (int i = tid; i < DHD * VT_STRIDE / 8; i += 256)
        *(uint4*)&vT[i * 8] = make_uint4(0u, 0u, 0u, 0u);
    __syncthreads();                                        // vT zeroed, K resident
    for (int c = tid; c < 592 * 8; c += 256) {             // V -> V^T scatter
        int j = c >> 3, part = c & 7;
        if (j < NN) {
            uint4 v = *(const uint4*)&qkv[(rowbase + j) * NQKV + 2 * DD + h * DHD + part * 8];
            const u16* pv = (const u16*)&v;
#pragma unroll
            for (int e = 0; e < 8; ++e) vT[(part * 8 + e) * VT_STRIDE + j] = pv[e];
        }
    }
    __syncthreads();

    // ---- per-wave i-tiles ----
    for (int it = wave; it < JT_TOT; it += 8) {
        const int i = it * 16 + lr;
        const bool ivalid = (i < NN);

        Frag qf[2];                                         // B operand: Q^T (q pre-scaled)
        if (ivalid) {
            const u16* qp = &qkv[(rowbase + i) * NQKV + h * DHD];
#pragma unroll
            for (int ks = 0; ks < 2; ++ks) {
                qf[ks].u[0] = *(const uint4*)(qp + ks * 32 + half * 8);
                qf[ks].u[1] = *(const uint4*)(qp + ks * 32 + half * 8 + 16);
            }
        } else {
#pragma unroll
            for (int ks = 0; ks < 2; ++ks) {
                qf[ks].u[0] = make_uint4(0u, 0u, 0u, 0u);
                qf[ks].u[1] = make_uint4(0u, 0u, 0u, 0u);
            }
        }

        // pass 1: row max over valid j
        float mx = -3.0e38f;
        for (int jt = 0; jt < JT_TOT; ++jt) {
            v8f st = st_tile(kl, qf, jt, lr, half);
#pragma unroll
            for (int r = 0; r < 8; ++r) {
                int j = jt * 16 + half * 8 + r;
                if (j < NN) mx = fmaxf(mx, st[r]);
            }
        }
        mx = fmaxf(mx, __shfl_xor(mx, 16, 32));

        // pass 2: exp-sum + two WMMA accumulators (softmax part / bias part)
        float l = 0.0f;
        v8f accS[4], accB[4];
#pragma unroll
        for (int md = 0; md < 4; ++md) { accS[md] = (v8f){}; accB[md] = (v8f){}; }

        for (int jp = 0; jp < JPAIRS; ++jp) {
            Frag pfrag, bfrag;
#pragma unroll
            for (int tsub = 0; tsub < 2; ++tsub) {
                int jt = jp * 2 + tsub;
                v8f st = (v8f){};
                if (jt < JT_TOT) st = st_tile(kl, qf, jt, lr, half);
#pragma unroll
                for (int r = 0; r < 8; ++r) {
                    int j = jt * 16 + half * 8 + r;
                    bool jv = (j < NN);
                    float p = jv ? __expf(st[r] - mx) : 0.0f;
                    l += p;
                    float bb = (jv && ivalid) ? rbs[NN - 1 + j - i] : 0.0f;
                    pfrag.s[tsub * 8 + r] = f2bf(p);
                    bfrag.s[tsub * 8 + r] = f2bf(bb);
                }
            }
#pragma unroll
            for (int md = 0; md < 4; ++md) {
                Frag vf;                                    // A operand: V^T tile
                const u16* p = &vT[(md * 16 + lr) * VT_STRIDE + jp * 32 + half * 8];
                vf.u[0] = *(const uint4*)p;
                vf.u[1] = *(const uint4*)(p + 16);
                accS[md] = __builtin_amdgcn_wmma_f32_16x16x32_bf16(
                    false, vf.v, false, pfrag.v, (short)0, accS[md], false, false);
                accB[md] = __builtin_amdgcn_wmma_f32_16x16x32_bf16(
                    false, vf.v, false, bfrag.v, (short)0, accB[md], false, false);
            }
        }
        l += __shfl_xor(l, 16, 32);
        float inv = 1.0f / l;

        if (ivalid) {
            u16* op = &aout[(rowbase + i) * DD + h * DHD];
#pragma unroll
            for (int md = 0; md < 4; ++md) {
                uint4 pk;
                u16* ps = (u16*)&pk;
#pragma unroll
                for (int r = 0; r < 8; ++r)
                    ps[r] = f2bf(accS[md][r] * inv + accB[md][r]);
                *(uint4*)(op + md * 16 + half * 8) = pk;    // O[i][d], d = md*16+half*8+r
            }
        }
    }
}

// ---------------------------------------------------------------------------
// Launch
// ---------------------------------------------------------------------------
extern "C" void kernel_launch(void* const* d_in, const int* in_sizes, int n_in,
                              void* d_out, int out_size, void* d_ws, size_t ws_size,
                              hipStream_t stream) {
    const float* x      = (const float*)d_in[0];
    const float* Wqk    = (const float*)d_in[1];
    const float* Wv     = (const float*)d_in[2];
    const float* Wout   = (const float*)d_in[3];
    const float* bout   = (const float*)d_in[4];
    const float* relb   = (const float*)d_in[5];
    const float* edelta = (const float*)d_in[6];
    const float* ealpha = (const float*)d_in[7];
    const float* ebeta  = (const float*)d_in[8];
    const float* egamma = (const float*)d_in[9];
    const float* eomega = (const float*)d_in[10];
    float* out = (float*)d_out;

    char* ws = (char*)d_ws;
    // workspace layout (~115.5 MB total)
    u16*   wall = (u16*)(ws + 0);                       //  1.57 MB
    u16*   wout = (u16*)(ws + 1572864);                 //  0.52 MB
    u16*   xm   = (u16*)(ws + 2097152);                 // 18.9 MB
    u16*   qkv  = (u16*)(ws + 21004288);                // 56.7 MB
    float* yfwd = (float*)(ws + 77725696);              // 37.7 MB (EMA scratch)
    u16*   aout = (u16*)(ws + 77725696);                // reused after EMA consumed

    prep_wall_kernel<<<(DD * NQKV + 255) / 256, 256, 0, stream>>>(Wqk, Wv, wall);
    prep_wout_kernel<<<(DD * DD + 255) / 256, 256, 0, stream>>>(Wout, wout);
    prep_cls_kernel<<<(BB * DD + 255) / 256, 256, 0, stream>>>(x, xm);

    ema_fwd_kernel<<<(BB * DD) / 256, 256, 0, stream>>>(x, edelta, ealpha, ebeta, egamma, yfwd);
    ema_bwd_kernel<<<(BB * DD) / 256, 256, 0, stream>>>(x, edelta, ealpha, ebeta, egamma,
                                                        eomega, yfwd, xm);

    dim3 gq((MROWS + 127) / 128, NQKV / 128);
    gemm_wmma_kernel<0><<<gq, 256, 0, stream>>>(xm, wall, qkv, nullptr, nullptr,
                                                MROWS, NQKV, DD);

    size_t smem = (size_t)592 * DHD * 2 + (size_t)DHD * VT_STRIDE * 2 + (2 * NN - 1) * 4;
    hipFuncSetAttribute((const void*)attention_kernel,
                        hipFuncAttributeMaxDynamicSharedMemorySize, (int)smem);
    attention_kernel<<<BB * HH, 256, smem, stream>>>(qkv, relb, aout);

    dim3 go((MROWS + 127) / 128, DD / 128);
    gemm_wmma_kernel<1><<<go, 256, 0, stream>>>(aout, wout, nullptr, out, bout,
                                                MROWS, DD, 0);
}
